// Encoder_Ankle_81810537054560
// MI455X (gfx1250) — compile-verified, matrix operand order checked by reading, not run on hardware
//
#include <hip/hip_runtime.h>
#include <cstdint>
#include <cstddef>

// ---------------------------------------------------------------------------
// Types for CDNA5 WMMA (wave32): A/B are 16 bf16 per lane, C/D are 8 f32.
// ---------------------------------------------------------------------------
typedef __bf16 bf16;
typedef __attribute__((ext_vector_type(16))) __bf16 v16bf;
typedef __attribute__((ext_vector_type(8)))  __bf16 v8bf;
typedef __attribute__((ext_vector_type(8)))  float  v8f;
typedef __attribute__((ext_vector_type(4)))  unsigned u32x4;
typedef __attribute__((ext_vector_type(8)))  int      i32x8;
typedef __attribute__((ext_vector_type(4)))  int      i32x4;

#define DI __device__ __forceinline__

namespace {
constexpr int Bn = 128;    // batch
constexpr int Tn = 127;    // time steps
constexpr int Fn = 128;    // features
constexpr int Hn = 512;    // hidden
constexpr int Tp = 128;    // T padded to multiple of 16
constexpr int G4 = 4 * Hn; // 2048 gate rows
constexpr int MF = Bn * Fn;           // 16384 (rows of inp_proj GEMM)
constexpr int GP_COLS = 64;           // gate columns handled per workgroup
constexpr int SG_LD = 65;             // padded LDS stride for gate tile (banks)
constexpr size_t SM_WIH = (size_t)GP_COLS * Fn * sizeof(bf16);   // 16 KB
constexpr size_t SM_WHH = (size_t)GP_COLS * Hn * sizeof(bf16);   // 64 KB
constexpr size_t SM_G   = (size_t)Bn * SG_LD * sizeof(float);    // 32.5 KB
constexpr size_t SM_TOTAL = SM_WIH + SM_WHH + SM_G;              // ~113 KB of 320 KB/WGP
}

// ---------------------------------------------------------------------------
// TDM availability (probe-verified builtin; arity differs across toolchains).
// ---------------------------------------------------------------------------
#if defined(__has_builtin)
#if __has_builtin(__builtin_amdgcn_tensor_load_to_lds)
#define HAVE_TDM 1
#endif
#if __has_builtin(__builtin_amdgcn_s_wait_tensorcnt)
#define HAVE_TENSORCNT 1
#endif
#endif

#ifdef HAVE_TDM
// Issue a 1-D TDM transfer of nElems bf16 elements (contiguous) global -> LDS.
// D# packed per CDNA5 ISA 8.3/8.4: group0 = {flags, lds_addr, global_addr, type},
// group1 = {data_size, tensor_dim0, tile_dim0, stride}. Groups 2/3 unused (2-D max).
DI void tdm_load_1d(unsigned lds_addr, const void* gptr, unsigned nElems) {
    const unsigned long long ga = (unsigned long long)gptr;
    u32x4 g0;
    g0[0] = 1u;                                   // count=1 valid descriptor
    g0[1] = lds_addr;                             // LDS byte address
    g0[2] = (unsigned)(ga & 0xffffffffu);         // global_addr[31:0]
    g0[3] = (unsigned)((ga >> 32) & 0x1ffffffu)   // global_addr[56:32]
          | (2u << 30);                           // type = 2 ("image")
    i32x8 g1;
    g1[0] = (int)(1u << 16);                      // data_size = 1 (2 bytes)
    g1[1] = (int)((nElems & 0xffffu) << 16);      // tensor_dim0[15:0]  @ bits 63:48
    g1[2] = (int)((nElems >> 16) & 0xffffu);      // tensor_dim0[31:16] @ bits 79:64
    g1[3] = (int)((nElems & 0xffffu) << 16);      // tile_dim0 @ bits 127:112
    g1[4] = 0;                                    // tile_dim1 = tile_dim2 = 0
    g1[5] = (int)nElems;                          // tensor_dim0_stride[31:0]
    g1[6] = 0;
    g1[7] = 0;
    i32x4 z4 = {0, 0, 0, 0};
#if __clang_major__ >= 23
    i32x8 z8 = {0, 0, 0, 0, 0, 0, 0, 0};
    __builtin_amdgcn_tensor_load_to_lds(g0, g1, z4, z4, z8, 0);
#else
    __builtin_amdgcn_tensor_load_to_lds(g0, g1, z4, z4, 0);
#endif
}

DI void tdm_wait0() {
#ifdef HAVE_TENSORCNT
    __builtin_amdgcn_s_wait_tensorcnt(0);
#else
    asm volatile("s_wait_tensorcnt 0" ::: "memory");
#endif
}
#endif // HAVE_TDM

// ---------------------------------------------------------------------------
// 16x32 bf16 operand fragment (ISA 7.12.2) from row-major matrix, ld elems.
// Lane l<16 : row = l,     K = {kb+0..7,  kb+16..23}
// Lane l>=16: row = l-16,  K = {kb+8..15, kb+24..31}   -> two 16-byte loads.
// Works for A (X, MxK) and B (W, NxK): every GEMM here is X @ W^T.
// ---------------------------------------------------------------------------
DI v16bf load_frag(const bf16* __restrict__ base, int ld, int row0, int kb, int lane) {
    const int half = (lane >> 4) & 1;
    const int r    = lane & 15;
    const bf16* p  = base + (size_t)(row0 + r) * (size_t)ld + kb + half * 8;
    v8bf lo = *(const v8bf*)(p);
    v8bf hi = *(const v8bf*)(p + 16);
    v16bf out;
#pragma unroll
    for (int i = 0; i < 8; ++i) { out[i] = lo[i]; out[i + 8] = hi[i]; }
    return out;
}

DI v8f wmma_bf16(v16bf a, v16bf b, v8f c) {
    return __builtin_amdgcn_wmma_f32_16x16x32_bf16(false, a, false, b, (short)0, c,
                                                   false, false);
}

DI float sigmoidf_(float x) { return 1.0f / (1.0f + __expf(-x)); }

// ---------------------------------------------------------------------------
// One-time prep kernels
// ---------------------------------------------------------------------------

// Split W1 (127 x 1151) into bf16 W1_h/W1_c (128x512) and W1_x (128x128), zero-pad.
__global__ void k_prep_w1(const float* __restrict__ W1,
                          bf16* __restrict__ Wh, bf16* __restrict__ Wc,
                          bf16* __restrict__ Wx) {
    const int u   = blockIdx.x;
    const int tid = threadIdx.x;
    const bool valid = (u < Tn);
    const float* row = W1 + (size_t)u * (2 * Hn + Tn);
    for (int c = tid; c < Hn; c += 128) {
        Wh[(size_t)u * Hn + c] = valid ? (bf16)row[c]      : (bf16)0.0f;
        Wc[(size_t)u * Hn + c] = valid ? (bf16)row[Hn + c] : (bf16)0.0f;
    }
    const int c = tid;
    Wx[(size_t)u * Tp + c] = (valid && c < Tn) ? (bf16)row[2 * Hn + c] : (bf16)0.0f;
}

// Gate-permuted bf16 weights: row g' = 4*j + gate  <-  row g = gate*H + j.
// Also fused bias biasp[g'] = bih[g] + bhh[g].
__global__ void k_prep_gatew(const float* __restrict__ Wih, const float* __restrict__ Whh,
                             const float* __restrict__ bih, const float* __restrict__ bhh,
                             bf16* __restrict__ Wihp, bf16* __restrict__ Whhp,
                             float* __restrict__ biasp) {
    const int i = blockIdx.x * blockDim.x + threadIdx.x;
    if (i < G4 * Hn) {
        const int gp = i / Hn, k = i - gp * Hn;
        const int j = gp >> 2, gate = gp & 3;
        Whhp[(size_t)gp * Hn + k] = (bf16)Whh[(size_t)(gate * Hn + j) * Hn + k];
    }
    if (i < G4 * Fn) {
        const int gp = i / Fn, k = i - gp * Fn;
        const int j = gp >> 2, gate = gp & 3;
        Wihp[(size_t)gp * Fn + k] = (bf16)Wih[(size_t)(gate * Hn + j) * Fn + k];
    }
    if (i < G4) {
        const int j = i >> 2, gate = i & 3;
        biasp[i] = bih[gate * Hn + j] + bhh[gate * Hn + j];
    }
}

// Xp_bf[(b*F+f)][tp] = bf16(x[b][t][f]), t padded to 128.
__global__ void k_prep_xp(const float* __restrict__ x, bf16* __restrict__ Xp) {
    const int i = blockIdx.x * blockDim.x + threadIdx.x;
    if (i >= MF * Tp) return;
    const int t   = i & (Tp - 1);
    const int bfi = i >> 7;
    const int f   = bfi & (Fn - 1);
    const int b   = bfi >> 7;
    Xp[i] = (t < Tn) ? (bf16)x[((size_t)b * Tn + t) * Fn + f] : (bf16)0.0f;
}

// Zero h (both buffers), c (bf16) and c (f32).
__global__ void k_init_state(bf16* __restrict__ h0, bf16* __restrict__ h1,
                             bf16* __restrict__ cB, float* __restrict__ cF) {
    const int i = blockIdx.x * blockDim.x + threadIdx.x;
    if (i >= Bn * Hn) return;
    h0[i] = (bf16)0.0f; h1[i] = (bf16)0.0f; cB[i] = (bf16)0.0f; cF[i] = 0.0f;
}

// inp_proj^T[u][(b*F+f)] = (Xp @ Wx^T)[bf][u] + b1[u].  M=16384, N=128, K=128.
// Stored transposed so the per-step attention loop reads coalesced.
__global__ void k_inp_proj(const bf16* __restrict__ Xp, const bf16* __restrict__ Wx,
                           const float* __restrict__ b1, float* __restrict__ outT) {
    const int wave = (int)((blockIdx.x * blockDim.x + threadIdx.x) >> 5);
    const int lane = threadIdx.x & 31;
    const int mt = wave >> 3;          // 0..1023
    const int nt = wave & 7;           // 0..7
    const int m0 = mt * 16, n0 = nt * 16;
    v8f acc = {};
#pragma unroll
    for (int kb = 0; kb < Tp; kb += 32) {
        v16bf a = load_frag(Xp, Tp, m0, kb, lane);
        v16bf b = load_frag(Wx, Tp, n0, kb, lane);
        acc = wmma_bf16(a, b, acc);
    }
    const int half = lane >> 4, col = lane & 15;
    const int u = n0 + col;
    const float bias = (u < Tn) ? b1[u] : 0.0f;
#pragma unroll
    for (int j = 0; j < 8; ++j) {
        const int m = m0 + j + half * 8;
        outT[(size_t)u * MF + m] = acc[j] + bias;
    }
}

// ---------------------------------------------------------------------------
// Per-step kernels
// ---------------------------------------------------------------------------

// state_proj[b][u] = h @ Wh^T + c @ Wc^T.  M=128, N=128, K=512 (x2).
__global__ void k_state_proj(const bf16* __restrict__ hB, const bf16* __restrict__ cB,
                             const bf16* __restrict__ Wh, const bf16* __restrict__ Wc,
                             float* __restrict__ sp) {
    const int wave = (int)((blockIdx.x * blockDim.x + threadIdx.x) >> 5); // 0..63
    const int lane = threadIdx.x & 31;
    const int mt = wave >> 3, nt = wave & 7;
    const int m0 = mt * 16, n0 = nt * 16;
    v8f acc = {};
#pragma unroll 4
    for (int kb = 0; kb < Hn; kb += 32) {
        v16bf a = load_frag(hB, Hn, m0, kb, lane);
        v16bf b = load_frag(Wh, Hn, n0, kb, lane);
        acc = wmma_bf16(a, b, acc);
    }
#pragma unroll 4
    for (int kb = 0; kb < Hn; kb += 32) {
        v16bf a = load_frag(cB, Hn, m0, kb, lane);
        v16bf b = load_frag(Wc, Hn, n0, kb, lane);
        acc = wmma_bf16(a, b, acc);
    }
    const int half = lane >> 4, col = lane & 15;
#pragma unroll
    for (int j = 0; j < 8; ++j)
        sp[(size_t)(m0 + j + half * 8) * Tp + n0 + col] = acc[j];
}

// score -> softmax over F -> attn, xw. One block per batch, one thread per f.
__global__ void k_attn(const float* __restrict__ ipT, const float* __restrict__ sp,
                       const float* __restrict__ W2, const float* __restrict__ b2,
                       const float* __restrict__ x, int t,
                       float* __restrict__ out_attn, float* __restrict__ out_xw,
                       bf16* __restrict__ xwB) {
    const int b = blockIdx.x;
    const int f = threadIdx.x;
    const float* ipc = ipT + (size_t)b * Fn + f;   // column (b,f); stride MF over u
    const float* spr = sp + (size_t)b * Tp;
    float s = b2[0];
#pragma unroll 4
    for (int u = 0; u < Tn; ++u)
        s += tanhf(ipc[(size_t)u * MF] + spr[u]) * W2[u];

    __shared__ float red[Fn];
    red[f] = s; __syncthreads();
    for (int off = Fn / 2; off > 0; off >>= 1) {
        if (f < off) red[f] = fmaxf(red[f], red[f + off]);
        __syncthreads();
    }
    const float mx = red[0]; __syncthreads();
    const float e = __expf(s - mx);
    red[f] = e; __syncthreads();
    for (int off = Fn / 2; off > 0; off >>= 1) {
        if (f < off) red[f] += red[f + off];
        __syncthreads();
    }
    const float attn = e / red[0];
    const float xv = x[((size_t)b * Tn + t) * Fn + f];
    const float xw = attn * xv;
    out_attn[((size_t)b * Tn + t) * Fn + f] = attn;
    out_xw [((size_t)b * Tn + t) * Fn + f] = xw;
    xwB[b * Fn + f] = (bf16)xw;
}

// Fused gates GEMM + LSTM pointwise. Each block owns 64 permuted gate columns
// (= 16 hidden units x 4 gates) for all 128 batch rows, K fully reduced:
//   1) stage Wih/Whh panels into LDS (TDM async-DMA if available),
//   2) 4 waves x (1 N-tile x 8 M-tiles) WMMA with LDS-resident B fragments,
//   3) gate tile -> LDS (padded stride), LSTM pointwise, write h/c.
__global__ void k_gates_fused(const bf16* __restrict__ xwB, const bf16* __restrict__ Wihp,
                              const bf16* __restrict__ hIn,  const bf16* __restrict__ Whhp,
                              const float* __restrict__ biasp,
                              float* __restrict__ cF, bf16* __restrict__ hOut,
                              bf16* __restrict__ cB, float* __restrict__ out_h, int t) {
    extern __shared__ char smem[];
    bf16*  sWih = (bf16*)smem;
    bf16*  sWhh = (bf16*)(smem + SM_WIH);
    float* sG   = (float*)(smem + SM_WIH + SM_WHH);

    const int tid  = threadIdx.x;
    const int wave = tid >> 5;
    const int lane = tid & 31;
    const int n0blk = blockIdx.x * GP_COLS;   // global permuted-gate column base

#ifdef HAVE_TDM
    if (tid < 32) {   // wave 0 drives the Tensor Data Mover for both panels
        tdm_load_1d((unsigned)(size_t)sWih, Wihp + (size_t)n0blk * Fn, GP_COLS * Fn);
        tdm_load_1d((unsigned)(size_t)sWhh, Whhp + (size_t)n0blk * Hn, GP_COLS * Hn);
        tdm_wait0();
    }
#else
    {   // cooperative vector copy fallback (panels are contiguous row ranges)
        const uint4* gi = (const uint4*)(Wihp + (size_t)n0blk * Fn);
        uint4* di = (uint4*)sWih;
        for (int i = tid; i < GP_COLS * Fn / 8; i += 128) di[i] = gi[i];
        const uint4* gh = (const uint4*)(Whhp + (size_t)n0blk * Hn);
        uint4* dh = (uint4*)sWhh;
        for (int i = tid; i < GP_COLS * Hn / 8; i += 128) dh[i] = gh[i];
    }
#endif
    __syncthreads();

    // GEMM: wave handles N-tile (wave*16 local) across all 8 M-tiles.
    v8f acc[8];
#pragma unroll
    for (int m = 0; m < 8; ++m) acc[m] = (v8f){};

#pragma unroll
    for (int kb = 0; kb < Fn; kb += 32) {
        v16bf b = load_frag(sWih, Fn, wave * 16, kb, lane);
#pragma unroll
        for (int m = 0; m < 8; ++m) {
            v16bf a = load_frag(xwB, Fn, m * 16, kb, lane);
            acc[m] = wmma_bf16(a, b, acc[m]);
        }
    }
#pragma unroll 2
    for (int kb = 0; kb < Hn; kb += 32) {
        v16bf b = load_frag(sWhh, Hn, wave * 16, kb, lane);
#pragma unroll
        for (int m = 0; m < 8; ++m) {
            v16bf a = load_frag(hIn, Hn, m * 16, kb, lane);
            acc[m] = wmma_bf16(a, b, acc[m]);
        }
    }

    // Gate tile (+bias) into padded LDS.
    const int half = lane >> 4, col = lane & 15;
    const int colL = wave * 16 + col;                 // local permuted column 0..63
    const float bias = biasp[n0blk + colL];
#pragma unroll
    for (int m = 0; m < 8; ++m)
#pragma unroll
        for (int j = 0; j < 8; ++j)
            sG[(size_t)(m * 16 + j + half * 8) * SG_LD + colL] = acc[m][j] + bias;
    __syncthreads();

    // LSTM pointwise: this block owns hidden units j = blockIdx*16 .. +15.
    for (int item = tid; item < Bn * 16; item += 128) {
        const int b  = item >> 4;
        const int jj = item & 15;
        const int j  = blockIdx.x * 16 + jj;
        const float* g = sG + (size_t)b * SG_LD + jj * 4;   // [i, f, g, o]
        const float ig = sigmoidf_(g[0]);
        const float fg = sigmoidf_(g[1]);
        const float gg = tanhf(g[2]);
        const float og = sigmoidf_(g[3]);
        const float c2 = fg * cF[(size_t)b * Hn + j] + ig * gg;
        const float h2 = og * tanhf(c2);
        cF[(size_t)b * Hn + j]   = c2;
        hOut[(size_t)b * Hn + j] = (bf16)h2;
        cB[(size_t)b * Hn + j]   = (bf16)c2;
        out_h[((size_t)b * Tn + t) * Hn + j] = h2;
    }
}

// ---------------------------------------------------------------------------
// Launch
// ---------------------------------------------------------------------------
extern "C" void kernel_launch(void* const* d_in, const int* in_sizes, int n_in,
                              void* d_out, int out_size, void* d_ws, size_t ws_size,
                              hipStream_t stream) {
    (void)in_sizes; (void)n_in; (void)out_size; (void)ws_size;

    const float* x    = (const float*)d_in[0];
    const float* W1   = (const float*)d_in[1];
    const float* b1   = (const float*)d_in[2];
    const float* W2   = (const float*)d_in[3];
    const float* b2   = (const float*)d_in[4];
    const float* Wih  = (const float*)d_in[5];
    const float* Whh  = (const float*)d_in[6];
    const float* bih  = (const float*)d_in[7];
    const float* bhh  = (const float*)d_in[8];

    float* out_attn = (float*)d_out;
    float* out_h    = out_attn + (size_t)Bn * Tn * Fn;
    float* out_xw   = out_h    + (size_t)Bn * Tn * Hn;

    char* basep = (char*)d_ws;
    size_t off = 0;
    auto take = [&](size_t bytes) -> char* {
        char* r = basep + off;
        off += (bytes + 255) & ~(size_t)255;
        return r;
    };
    bf16*  Xp    = (bf16*) take((size_t)MF * Tp * sizeof(bf16));
    bf16*  WxB   = (bf16*) take((size_t)Tp * Tp * sizeof(bf16));
    bf16*  WhB   = (bf16*) take((size_t)Tp * Hn * sizeof(bf16));
    bf16*  WcB   = (bf16*) take((size_t)Tp * Hn * sizeof(bf16));
    bf16*  Wihp  = (bf16*) take((size_t)G4 * Fn * sizeof(bf16));
    bf16*  Whhp  = (bf16*) take((size_t)G4 * Hn * sizeof(bf16));
    float* biasp = (float*)take((size_t)G4 * sizeof(float));
    float* ipT   = (float*)take((size_t)MF * Tp * sizeof(float));
    float* sp    = (float*)take((size_t)Bn * Tp * sizeof(float));
    bf16*  hBuf0 = (bf16*) take((size_t)Bn * Hn * sizeof(bf16));
    bf16*  hBuf1 = (bf16*) take((size_t)Bn * Hn * sizeof(bf16));
    bf16*  cB    = (bf16*) take((size_t)Bn * Hn * sizeof(bf16));
    float* cF    = (float*)take((size_t)Bn * Hn * sizeof(float));
    bf16*  xwB   = (bf16*) take((size_t)Bn * Fn * sizeof(bf16));

    // ---- one-time prep ----
    k_prep_w1   <<<Tp, 128, 0, stream>>>(W1, WhB, WcB, WxB);
    k_prep_gatew<<<(G4 * Hn + 255) / 256, 256, 0, stream>>>(Wih, Whh, bih, bhh,
                                                            Wihp, Whhp, biasp);
    k_prep_xp   <<<(MF * Tp + 255) / 256, 256, 0, stream>>>(x, Xp);
    k_init_state<<<(Bn * Hn + 255) / 256, 256, 0, stream>>>(hBuf0, hBuf1, cB, cF);
    k_inp_proj  <<<8192 / 4, 128, 0, stream>>>(Xp, WxB, b1, ipT);

    // ---- serial scan over time (h double-buffered across the fused kernel) ----
    bf16* hCur = hBuf0;
    bf16* hNxt = hBuf1;
    for (int t = 0; t < Tn; ++t) {
        k_state_proj <<<16, 128, 0, stream>>>(hCur, cB, WhB, WcB, sp);
        k_attn       <<<Bn, Fn, 0, stream>>>(ipT, sp, W2, b2, x, t,
                                             out_attn, out_xw, xwB);
        k_gates_fused<<<G4 / GP_COLS, 128, (unsigned)SM_TOTAL, stream>>>(
            xwB, Wihp, hCur, Whhp, biasp, cF, hNxt, cB, out_h, t);
        bf16* tmp = hCur; hCur = hNxt; hNxt = tmp;
    }
}
